// CNN_62139586839005
// MI455X (gfx1250) — compile-verified, matrix operand order checked by reading, not run on hardware
//
#include <hip/hip_runtime.h>
#include <hip/hip_bf16.h>
#include <stdint.h>

// ---------------------------------------------------------------------------
// Sparse voxel CNN for MI455X (gfx1250, wave32).
// Stride-2 convs run as im2col GEMMs through v_wmma_f32_16x16x32_bf16
// (f32 accumulate); BN+ReLU+mask fused into the A-fragment gather; dead
// 16-voxel tiles skipped wave-uniformly (EXEC stays all-ones for WMMA).
// ---------------------------------------------------------------------------

typedef __attribute__((ext_vector_type(16))) __bf16 v16bf;
typedef __attribute__((ext_vector_type(8)))  float  v8f;

#define SGRID 128
#define NB    2
#define NPTS  60000
#define NEGV  (-3e38f)

__device__ __forceinline__ __bf16 f2bf(float f) {
    union { float f; uint32_t u; } v; v.f = f;
    uint32_t r = v.u + 0x7FFFu + ((v.u >> 16) & 1u);   // round-to-nearest-even
    union { uint16_t s; __bf16 b; } o; o.s = (uint16_t)(r >> 16);
    return o.b;
}

// ---------------------------------------------------------------------------
__global__ void zero_kernel(float* __restrict__ p, size_t n) {
    size_t i = (size_t)blockIdx.x * blockDim.x + threadIdx.x;
    size_t st = (size_t)gridDim.x * blockDim.x;
    for (; i < n; i += st) p[i] = 0.f;
}

__global__ void scatter_kernel(const float* __restrict__ pc,
                               float* __restrict__ grid, float* __restrict__ mask) {
    int i = blockIdx.x * blockDim.x + threadIdx.x;
    if (i >= NPTS) return;
    const float* r = pc + (size_t)i * 5;
    int b = (int)r[0], x = (int)r[1], y = (int)r[2], z = (int)r[3];
    size_t idx = (((size_t)b * SGRID + x) * SGRID + y) * SGRID + z;
    atomicAdd(&grid[idx], r[4]);
    mask[idx] = 1.0f;
}

__global__ void coords_kernel(const float* __restrict__ pc, float* __restrict__ out) {
    int i = blockIdx.x * blockDim.x + threadIdx.x;
    if (i >= NPTS * 4) return;
    out[i] = pc[(size_t)(i >> 2) * 5 + (i & 3)];
}

// ---------------------------------------------------------------------------
// 3x3x3 SAME conv, 1->8 channels, output masked. Sparse: skip math when
// the output voxel is unoccupied (still writes zeros, one pass).
__global__ void conv_sub_kernel(const float* __restrict__ grid,
                                const float* __restrict__ mask,
                                const float* __restrict__ wsub,
                                float* __restrict__ h0) {
    __shared__ float w[216];
    int t = threadIdx.x;
    if (t < 216) w[t] = wsub[t];
    __syncthreads();
    const size_t V = (size_t)SGRID * SGRID * SGRID;
    size_t i = (size_t)blockIdx.x * blockDim.x + t;
    if (i >= (size_t)NB * V) return;
    int b = (int)(i / V);
    int v = (int)(i % V);
    float acc[8];
    #pragma unroll
    for (int c = 0; c < 8; ++c) acc[c] = 0.f;
    if (mask[i] > 0.f) {
        int z = v / (SGRID * SGRID);
        int y = (v / SGRID) % SGRID;
        int x = v % SGRID;
        for (int dz = 0; dz < 3; ++dz) {
            int zz = z + dz - 1; if (zz < 0 || zz >= SGRID) continue;
            for (int dy = 0; dy < 3; ++dy) {
                int yy = y + dy - 1; if (yy < 0 || yy >= SGRID) continue;
                for (int dx = 0; dx < 3; ++dx) {
                    int xx = x + dx - 1; if (xx < 0 || xx >= SGRID) continue;
                    float g = grid[(size_t)b * V + ((size_t)zz * SGRID + yy) * SGRID + xx];
                    int tap = (dz * 3 + dy) * 3 + dx;
                    #pragma unroll
                    for (int c = 0; c < 8; ++c) acc[c] += g * w[c * 27 + tap];
                }
            }
        }
    }
    #pragma unroll
    for (int c = 0; c < 8; ++c)
        h0[(size_t)(b * 8 + c) * V + v] = acc[c];
}

// ---------------------------------------------------------------------------
// Fused BN statistics: h is already masked, so sum/sumsq over the whole
// tensor equal the masked sums. One pass, block-reduced, atomic finish.
__global__ void bn_stats_kernel(const float* __restrict__ h, float* __restrict__ sums,
                                int C, int V) {
    int c = blockIdx.y;
    float s = 0.f, s2 = 0.f;
    for (int b = 0; b < NB; ++b) {
        const float* p = h + (size_t)(b * C + c) * V;
        for (int i = blockIdx.x * blockDim.x + threadIdx.x; i < V;
             i += gridDim.x * blockDim.x) {
            float x = p[i]; s += x; s2 += x * x;
        }
    }
    __shared__ float ls[256], ls2[256];
    int t = threadIdx.x;
    ls[t] = s; ls2[t] = s2;
    __syncthreads();
    for (int o = 128; o > 0; o >>= 1) {
        if (t < o) { ls[t] += ls[t + o]; ls2[t] += ls2[t + o]; }
        __syncthreads();
    }
    if (t == 0) { atomicAdd(&sums[c], ls[0]); atomicAdd(&sums[C + c], ls2[0]); }
}

__global__ void sum_kernel(const float* __restrict__ p, size_t n, float* __restrict__ out) {
    float s = 0.f;
    for (size_t i = (size_t)blockIdx.x * blockDim.x + threadIdx.x; i < n;
         i += (size_t)gridDim.x * blockDim.x) s += p[i];
    __shared__ float ls[256];
    int t = threadIdx.x;
    ls[t] = s;
    __syncthreads();
    for (int o = 128; o > 0; o >>= 1) {
        if (t < o) ls[t] += ls[t + o];
        __syncthreads();
    }
    if (t == 0) atomicAdd(out, ls[0]);
}

// sums[0..C) = sum(x), sums[C..2C) = sum(x^2), sums[2C] = count(mask)
// stats[0..C) = mu, stats[C..2C) = gamma * rsqrt(var + eps)
__global__ void bn_finalize_kernel(const float* __restrict__ sums,
                                   const float* __restrict__ gamma,
                                   float* __restrict__ stats, int C) {
    int c = blockIdx.x * blockDim.x + threadIdx.x;
    if (c >= C) return;
    float n   = sums[2 * C];
    float mu  = sums[c] / n;
    float var = sums[C + c] / n - mu * mu;
    stats[c]     = mu;
    stats[C + c] = gamma[c] * rsqrtf(var + 1e-4f);
}

// ---------------------------------------------------------------------------
__global__ void pool_mask_kernel(const float* __restrict__ min_, float* __restrict__ mout,
                                 int NIN) {
    int NOUT = NIN >> 1;
    int VOUT = NOUT * NOUT * NOUT;
    int i = blockIdx.x * blockDim.x + threadIdx.x;
    if (i >= NB * VOUT) return;
    int b = i / VOUT, v = i % VOUT;
    int z = v / (NOUT * NOUT), y = (v / NOUT) % NOUT, x = v % NOUT;
    const float* p = min_ + (size_t)b * NIN * NIN * NIN +
                     ((size_t)(2 * z) * NIN + 2 * y) * NIN + 2 * x;
    float m = 0.f;
    #pragma unroll
    for (int t = 0; t < 8; ++t) {
        int off = ((((t >> 2) & 1) * NIN) + ((t >> 1) & 1)) * NIN + (t & 1);
        m = fmaxf(m, p[off]);
    }
    mout[i] = m;
}

// masked maxpool: where(mask_in>0, h, NEG) -> pool -> where(mask_out>0, ., 0)
__global__ void masked_pool_kernel(const float* __restrict__ hin,
                                   const float* __restrict__ min_,
                                   const float* __restrict__ mout,
                                   float* __restrict__ hout, int C, int NIN) {
    int NOUT = NIN >> 1;
    int VIN = NIN * NIN * NIN, VOUT = NOUT * NOUT * NOUT;
    int i = blockIdx.x * blockDim.x + threadIdx.x;
    if (i >= NB * C * VOUT) return;
    int v = i % VOUT; int bc = i / VOUT; int c = bc % C; int b = bc / C;
    int z = v / (NOUT * NOUT), y = (v / NOUT) % NOUT, x = v % NOUT;
    size_t sb = ((size_t)(2 * z) * NIN + 2 * y) * NIN + 2 * x;
    const float* ph = hin + (size_t)(b * C + c) * VIN + sb;
    const float* pm = min_ + (size_t)b * VIN + sb;
    float val = NEGV;
    #pragma unroll
    for (int t = 0; t < 8; ++t) {
        int off = ((((t >> 2) & 1) * NIN) + ((t >> 1) & 1)) * NIN + (t & 1);
        float mm = pm[off];
        val = fmaxf(val, (mm > 0.f) ? ph[off] : NEGV);
    }
    float mo = mout[(size_t)b * VOUT + v];
    hout[i] = (mo > 0.f) ? val : 0.f;
}

// ---------------------------------------------------------------------------
// Stride-2 2x2x2 conv as GEMM on the matrix pipe.
//   M = output voxels (16/tile), N = Cout (16/tile), K = Cin*8, 32 per WMMA.
// A fragment (16-bit A 16x32 ISA layout): lane m = lane&15, khalf = lane>>4;
// v[0..3] hold K = khalf*8 + 0..7 (one full input channel's 8 taps),
// v[4..7] hold K = 16 + khalf*8 + 0..7 (another channel's 8 taps).
// BN+ReLU+mask fused into the gather. Dead tiles skipped wave-uniformly.
template <int CIN, int COUT, int NIN>
__global__ __launch_bounds__(128) void conv_wmma_kernel(
    const float* __restrict__ hin, const float* __restrict__ min_,
    const float* __restrict__ mout, const float* __restrict__ w,
    const float* __restrict__ stats, const float* __restrict__ beta,
    float* __restrict__ hout) {
    constexpr int NOUT = NIN / 2;
    constexpr int VIN  = NIN * NIN * NIN;
    constexpr int VOUT = NOUT * NOUT * NOUT;

    const int lane  = threadIdx.x & 31;
    const int wv    = threadIdx.x >> 5;
    const int mtile = blockIdx.x * 4 + wv;
    const int ntile = blockIdx.y;
    const int n     = lane & 15;
    const int khalf = lane >> 4;
    const int co    = ntile * 16 + n;

    // Sparsity: if all 16 output voxels of this tile are masked out, emit
    // zeros and bail (uniform branch -> EXEC stays all-ones around WMMA).
    const int voxA = mtile * 16 + n;
    const unsigned long long alive = __ballot(mout[voxA] > 0.f);
    if (alive == 0ull) {
        #pragma unroll
        for (int r = 0; r < 8; ++r) {
            int vox = mtile * 16 + r + 8 * khalf;
            int bs = vox / VOUT, vloc = vox % VOUT;
            hout[(size_t)(bs * COUT + co) * VOUT + vloc] = 0.f;
        }
        return;
    }

    // Decode this lane's A-row output voxel -> input patch base.
    const int bA = voxA / VOUT;
    const int rA = voxA % VOUT;
    const int z = rA / (NOUT * NOUT), y = (rA / NOUT) % NOUT, x = rA % NOUT;
    const size_t sbase = ((size_t)(2 * z) * NIN + 2 * y) * NIN + 2 * x;

    int   off[8];
    float mk[8];
    const float* pm = min_ + (size_t)bA * VIN + sbase;
    #pragma unroll
    for (int t = 0; t < 8; ++t) {
        off[t] = ((((t >> 2) & 1) * NIN) + ((t >> 1) & 1)) * NIN + (t & 1);
        mk[t]  = pm[off[t]];
    }

    v8f acc = {};
    const float* wrow = w + (size_t)co * (CIN * 8);

    #pragma unroll
    for (int kk = 0; kk < CIN / 4; ++kk) {
        const int k0  = kk * 32;
        const int ci0 = kk * 4 + khalf;
        const int ci1 = kk * 4 + 2 + khalf;

        // ---- A fragment: two input channels x 8 taps, BN+ReLU+mask fused.
        v16bf a;
        {
            const float mu0 = stats[ci0], sc0 = stats[CIN + ci0], be0 = beta[ci0];
            const float* p0 = hin + (size_t)(bA * CIN + ci0) * VIN + sbase;
            #pragma unroll
            for (int t = 0; t < 8; ++t) {
                float xv = p0[off[t]];
                a[t] = f2bf(fmaxf((xv - mu0) * sc0 + be0, 0.f) * mk[t]);
            }
            const float mu1 = stats[ci1], sc1 = stats[CIN + ci1], be1 = beta[ci1];
            const float* p1 = hin + (size_t)(bA * CIN + ci1) * VIN + sbase;
            #pragma unroll
            for (int t = 0; t < 8; ++t) {
                float xv = p1[off[t]];
                a[8 + t] = f2bf(fmaxf((xv - mu1) * sc1 + be1, 0.f) * mk[t]);
            }
        }

        // ---- B fragment: column n of W for this K-step (two runs of 8).
        v16bf bf;
        {
            const float* wb = wrow + k0 + khalf * 8;
            #pragma unroll
            for (int j = 0; j < 8; ++j) bf[j] = f2bf(wb[j]);
            #pragma unroll
            for (int j = 0; j < 8; ++j) bf[8 + j] = f2bf(wb[16 + j]);
        }

        if (kk + 1 < CIN / 4)  // hint next channel planes into cache
            __builtin_prefetch(hin + (size_t)(bA * CIN + ci0 + 4) * VIN + sbase, 0, 1);

        acc = __builtin_amdgcn_wmma_f32_16x16x32_bf16(
            false, a, false, bf, (short)0, acc, false, false);
    }

    // ---- Store D (16x16 f32 layout: lane n = lane&15, VGPR r -> m=r+8*khalf),
    // applying the post-conv output mask.
    #pragma unroll
    for (int r = 0; r < 8; ++r) {
        int vox = mtile * 16 + r + 8 * khalf;
        int bs = vox / VOUT, vloc = vox % VOUT;
        hout[(size_t)(bs * COUT + co) * VOUT + vloc] = acc[r] * mout[vox];
    }
}

// ---------------------------------------------------------------------------
__global__ void fc_kernel(const float* __restrict__ h6, const float* __restrict__ Wc,
                          const float* __restrict__ bc, float* __restrict__ out) {
    int b = blockIdx.x, o = blockIdx.y;
    float s = 0.f;
    for (int j = threadIdx.x; j < 1024; j += blockDim.x) {
        float xv = fmaxf(h6[b * 1024 + j], 0.f);
        s += xv * Wc[o * 1024 + j];
    }
    __shared__ float ls[256];
    ls[threadIdx.x] = s;
    __syncthreads();
    for (int t = 128; t > 0; t >>= 1) {
        if (threadIdx.x < t) ls[threadIdx.x] += ls[threadIdx.x + t];
        __syncthreads();
    }
    if (threadIdx.x == 0) out[b * 2 + o] = ls[0] + bc[o];
}

// ---------------------------------------------------------------------------
static inline void launch_stage_prep(const float* hin, const float* min_, float* mout,
                                     const float* gamma, float* sums, float* stats,
                                     int CIN, int NIN, hipStream_t stream) {
    int NOUT = NIN / 2;
    int VIN = NIN * NIN * NIN, VOUT = NOUT * NOUT * NOUT;
    zero_kernel<<<1, 256, 0, stream>>>(sums, (size_t)(2 * CIN + 1));
    int chunks = VIN / (256 * 16);
    if (chunks < 1) chunks = 1;
    if (chunks > 256) chunks = 256;
    bn_stats_kernel<<<dim3(chunks, CIN), 256, 0, stream>>>(hin, sums, CIN, VIN);
    int cblocks = (2 * VIN + 255) / 256;
    if (cblocks > 1024) cblocks = 1024;
    sum_kernel<<<cblocks, 256, 0, stream>>>(min_, (size_t)2 * VIN, sums + 2 * CIN);
    bn_finalize_kernel<<<1, 64, 0, stream>>>(sums, gamma, stats, CIN);
    pool_mask_kernel<<<(2 * VOUT + 255) / 256, 256, 0, stream>>>(min_, mout, NIN);
}

extern "C" void kernel_launch(void* const* d_in, const int* in_sizes, int n_in,
                              void* d_out, int out_size, void* d_ws, size_t ws_size,
                              hipStream_t stream) {
    const float* pc   = (const float*)d_in[0];
    const float* wsub = (const float*)d_in[1];
    const float* w0   = (const float*)d_in[2];
    const float* w1   = (const float*)d_in[3];
    const float* w2   = (const float*)d_in[4];
    const float* w3   = (const float*)d_in[5];
    const float* g0   = (const float*)d_in[6];
    const float* g1   = (const float*)d_in[7];
    const float* g2   = (const float*)d_in[8];
    const float* g3   = (const float*)d_in[9];
    const float* be0  = (const float*)d_in[10];
    const float* be1  = (const float*)d_in[11];
    const float* be2  = (const float*)d_in[12];
    const float* be3  = (const float*)d_in[13];
    const float* Wc   = (const float*)d_in[14];
    const float* bc   = (const float*)d_in[15];
    float* out = (float*)d_out;
    (void)in_sizes; (void)n_in; (void)out_size; (void)ws_size;

    // ---- workspace carve-out (floats) --------------------------------------
    constexpr size_t V7 = 2097152;  // 128^3
    constexpr size_t V6 = 262144;   // 64^3
    constexpr size_t V5 = 32768;    // 32^3
    constexpr size_t V4 = 4096;     // 16^3
    constexpr size_t V3 = 512;      // 8^3
    constexpr size_t V2 = 64;       // 4^3
    constexpr size_t V1 = 8;        // 2^3
    float* ws    = (float*)d_ws;
    float* grid  = ws;                  // 2*V7
    float* mask0 = grid  + 2 * V7;      // 2*V7
    float* h0    = mask0 + 2 * V7;      // 16*V7
    float* mask1 = h0    + 16 * V7;     // 2*V6
    float* h1    = mask1 + 2 * V6;      // 32*V6
    float* mask2 = h1    + 32 * V6;     // 2*V5
    float* h2    = mask2 + 2 * V5;      // 64*V5
    float* mask3 = h2    + 64 * V5;     // 2*V4
    float* h3    = mask3 + 2 * V4;      // 128*V4
    float* mask4 = h3    + 128 * V4;    // 2*V3
    float* h4    = mask4 + 2 * V3;      // 256*V3
    float* mask5 = h4    + 256 * V3;    // 2*V2
    float* h5    = mask5 + 2 * V2;      // 256*V2
    float* mask6 = h5    + 256 * V2;    // 2*V1
    float* h6    = mask6 + 2 * V1;      // 256*V1
    float* sums  = h6    + 256 * V1;    // up to 2*64+1
    float* stats = sums  + 132;         // up to 128

    // ---- voxelize ----------------------------------------------------------
    zero_kernel<<<4096, 256, 0, stream>>>(grid, 4 * V7);  // grid + mask0 contiguous
    scatter_kernel<<<(NPTS + 255) / 256, 256, 0, stream>>>(pc, grid, mask0);

    // ---- submanifold 3x3x3 conv (1->8) -------------------------------------
    conv_sub_kernel<<<(int)((2 * V7 + 255) / 256), 256, 0, stream>>>(grid, mask0, wsub, h0);

    // ---- four BN+ReLU -> stride-2 WMMA conv stages -------------------------
    launch_stage_prep(h0, mask0, mask1, g0, sums, stats, 8, 128, stream);
    conv_wmma_kernel<8, 16, 128><<<dim3(8192, 1), 128, 0, stream>>>(
        h0, mask0, mask1, w0, stats, be0, h1);

    launch_stage_prep(h1, mask1, mask2, g1, sums, stats, 16, 64, stream);
    conv_wmma_kernel<16, 32, 64><<<dim3(1024, 2), 128, 0, stream>>>(
        h1, mask1, mask2, w1, stats, be1, h2);

    launch_stage_prep(h2, mask2, mask3, g2, sums, stats, 32, 32, stream);
    conv_wmma_kernel<32, 64, 32><<<dim3(128, 4), 128, 0, stream>>>(
        h2, mask2, mask3, w2, stats, be2, h3);

    launch_stage_prep(h3, mask3, mask4, g3, sums, stats, 64, 16, stream);
    conv_wmma_kernel<64, 128, 16><<<dim3(16, 8), 128, 0, stream>>>(
        h3, mask3, mask4, w3, stats, be3, h4);

    // ---- two masked max-pools ----------------------------------------------
    pool_mask_kernel<<<1, 256, 0, stream>>>(mask4, mask5, 8);
    masked_pool_kernel<<<(2 * 128 * 64 + 255) / 256, 256, 0, stream>>>(
        h4, mask4, mask5, h5, 128, 8);
    pool_mask_kernel<<<1, 256, 0, stream>>>(mask5, mask6, 4);
    masked_pool_kernel<<<(2 * 128 * 8 + 255) / 256, 256, 0, stream>>>(
        h5, mask5, mask6, h6, 128, 4);

    // ---- classifier + coords passthrough -----------------------------------
    fc_kernel<<<dim3(NB, 2), 256, 0, stream>>>(h6, Wc, bc, out + (size_t)NPTS * 4);
    coords_kernel<<<(NPTS * 4 + 255) / 256, 256, 0, stream>>>(pc, out);
}
